// Block_71494025609672
// MI455X (gfx1250) — matrix operand order repaired and checked by reading, not compiled, not run on hardware
//
#include <hip/hip_runtime.h>
#include <math.h>

// ---------------------------------------------------------------------------
// Transformer block for MI455X (gfx1250): bf16 WMMA everywhere.
//   D_MODEL=2048, N_HEADS=16, HEAD_DIM=128, B=2, T=2048, M=B*T=4096
// ---------------------------------------------------------------------------

typedef __attribute__((ext_vector_type(16))) __bf16 v16bf;
typedef __attribute__((ext_vector_type(8)))  __bf16 bf16x8;
typedef __attribute__((ext_vector_type(4)))  __bf16 bf16x4;
typedef __attribute__((ext_vector_type(8)))  float  v8f;

union BF16x16 { v16bf v; bf16x8 h[2]; };

__device__ inline __bf16 f2bf(float f) {
  union { float f; unsigned u; } a; a.f = f;
  unsigned r = a.u + 0x7FFFu + ((a.u >> 16) & 1u);   // round-to-nearest-even
  union { unsigned short s; __bf16 b; } o; o.s = (unsigned short)(r >> 16);
  return o.b;
}

__device__ inline v16bf load_frag2(const __bf16* p0, const __bf16* p1) {
  BF16x16 u;
  u.h[0] = *(const bf16x8*)p0;
  u.h[1] = *(const bf16x8*)p1;
  return u.v;
}

__device__ inline v8f vzero8() { v8f z; for (int e = 0; e < 8; ++e) z[e] = 0.0f; return z; }

// ---------------------------------------------------------------------------
// Generic GEMM:  C[M x N] = A[M x K] * Bw[K x N]   (Bw always f32 row-major;
// A either f32 or bf16 row-major).  M fixed = 4096.  K % 32 == 0, N % 128 == 0.
// Block tile 128x128, 256 threads = 8 waves (4 along M x 2 along N),
// wave tile 32x64 -> 2x4 v_wmma_f32_16x16x32_bf16 accumulators.
// EPI: 0 = QKV scatter, 1 = f32 store, 2 = bias+GELU->bf16, 3 = final residual
// ---------------------------------------------------------------------------
template <int EPI, typename AT>
__global__ __launch_bounds__(256) void gemm_wmma(
    const AT* __restrict__ A, const float* __restrict__ Bw,
    int N, int K,
    __bf16* __restrict__ oq, __bf16* __restrict__ okk, __bf16* __restrict__ ov,
    float* __restrict__ of32, __bf16* __restrict__ obf,
    const float* __restrict__ bias, const float* __restrict__ res1,
    const float* __restrict__ res2, float* __restrict__ ofinal)
{
  __shared__ __bf16 As[128][40];   // [m][k], +8 pad; row stride 80B (16B mult)
  __shared__ __bf16 Bs[128][40];   // [n][k] (transposed), +8 pad

  const int tid  = threadIdx.x;
  const int lane = tid & 31;
  const int wave = tid >> 5;
  const int wm   = wave & 3;       // wave M index (0..3)
  const int wn   = wave >> 2;      // wave N index (0..1)
  const int g    = lane >> 4;      // 16-lane half group
  const int ln   = lane & 15;

  const int bm = blockIdx.y * 128;
  const int bn = blockIdx.x * 128;

  v8f acc[2][4];
  for (int i = 0; i < 2; ++i)
    for (int j = 0; j < 4; ++j) acc[i][j] = vzero8();

  const int nkt = K >> 5;
  for (int kt = 0; kt < nkt; ++kt) {
    const int k0 = kt << 5;
    // ---- stage A tile (128 x 32) into LDS as bf16 ----
    for (int j = 0; j < 4; ++j) {
      const int p   = tid * 4 + j;       // 0..1023 quads
      const int row = p >> 3;            // /8
      const int col = (p & 7) << 2;      // 0..28
      const size_t gofs = (size_t)(bm + row) * K + k0 + col;
      if constexpr (sizeof(AT) == 4) {
        const float4 s = *(const float4*)(A + gofs);
        bf16x4 d; d[0] = f2bf(s.x); d[1] = f2bf(s.y); d[2] = f2bf(s.z); d[3] = f2bf(s.w);
        *(bf16x4*)&As[row][col] = d;
      } else {
        *(bf16x4*)&As[row][col] = *(const bf16x4*)(A + gofs);
      }
      if (kt + 1 < nkt) __builtin_prefetch((const void*)(A + gofs + 32), 0, 1);
    }
    // ---- stage B tile (32 x 128) into LDS transposed [n][k] as bf16 ----
    for (int j = 0; j < 4; ++j) {
      const int p    = tid * 4 + j;      // 0..1023 quads
      const int krow = p >> 5;           // 0..31
      const int ncol = (p & 31) << 2;    // 0..124
      const size_t gofs = (size_t)(k0 + krow) * N + bn + ncol;
      const float4 s = *(const float4*)(Bw + gofs);
      Bs[ncol + 0][krow] = f2bf(s.x);
      Bs[ncol + 1][krow] = f2bf(s.y);
      Bs[ncol + 2][krow] = f2bf(s.z);
      Bs[ncol + 3][krow] = f2bf(s.w);
      if (kt + 1 < nkt) __builtin_prefetch((const void*)(Bw + gofs + (size_t)32 * N), 0, 1);
    }
    __syncthreads();
    // ---- fragments + WMMA ----
    v16bf afr[2], bfr[4];
    for (int mf = 0; mf < 2; ++mf) {
      const int r = wm * 32 + mf * 16 + ln;                  // A: m = lane&15
      afr[mf] = load_frag2(&As[r][8 * g], &As[r][16 + 8 * g]);
    }
    for (int nf = 0; nf < 4; ++nf) {
      const int c = wn * 64 + nf * 16 + ln;                  // B: n = lane&15
      bfr[nf] = load_frag2(&Bs[c][8 * g], &Bs[c][16 + 8 * g]);
    }
    for (int mf = 0; mf < 2; ++mf)
      for (int nf = 0; nf < 4; ++nf)
        acc[mf][nf] = __builtin_amdgcn_wmma_f32_16x16x32_bf16(
            false, afr[mf], false, bfr[nf], (short)0, acc[mf][nf], false, false);
    __syncthreads();
  }

  // ---- epilogue:  C-frag mapping m = i + 8*g, n = lane&15 ----
  for (int mf = 0; mf < 2; ++mf) {
    for (int nf = 0; nf < 4; ++nf) {
      for (int i = 0; i < 8; ++i) {
        const int m = bm + wm * 32 + mf * 16 + i + 8 * g;
        const int n = bn + wn * 64 + nf * 16 + ln;
        const float vacc = acc[mf][nf][i];
        if constexpr (EPI == 0) {
          // QKV scatter into per-head [B][H][T][128] bf16 buffers
          const int sect   = n >> 11;       // /2048 -> 0:q 1:k 2:v
          const int within = n & 2047;
          const int h      = within >> 7;   // /128
          const int d      = within & 127;
          const int b      = m >> 11;       // /T
          const int t      = m & 2047;
          const size_t idx = ((size_t)(b * 16 + h) * 2048 + t) * 128 + d;
          const __bf16 bv = f2bf(vacc);
          if (sect == 0) oq[idx] = bv;
          else if (sect == 1) okk[idx] = bv;
          else ov[idx] = bv;
        } else if constexpr (EPI == 1) {
          of32[(size_t)m * N + n] = vacc;
        } else if constexpr (EPI == 2) {
          const float xg = vacc + bias[n];
          const float gl = 0.5f * xg * (1.0f + erff(xg * 0.70710678118654752f));
          obf[(size_t)m * N + n] = f2bf(gl);
        } else {
          const size_t idx = (size_t)m * N + n;
          ofinal[idx] = res1[idx] + res2[idx] + vacc + bias[n];
        }
      }
    }
  }
}

// ---------------------------------------------------------------------------
// Causal flash attention: grid = (B*H, T/64), 128 threads = 4 waves,
// each wave owns 16 query rows.  hd = 128.  All matmuls via WMMA bf16.
// ---------------------------------------------------------------------------
__global__ __launch_bounds__(128) void attn_wmma(
    const __bf16* __restrict__ Q, const __bf16* __restrict__ Kb,
    const __bf16* __restrict__ Vb, __bf16* __restrict__ O)
{
  __shared__ __bf16 Ks[64][136];    // [key][d], row stride 272B (16B mult)
  __shared__ __bf16 Vts[128][80];   // [d][key] transposed, row stride 160B
  __shared__ __bf16 Ps[4][16][80];  // per-wave P tile [q][key]

  const int tid  = threadIdx.x;
  const int lane = tid & 31;
  const int w    = tid >> 5;
  const int g    = lane >> 4;
  const int ln   = lane & 15;
  const int bh   = blockIdx.x;           // b*H + h
  const int q0   = blockIdx.y * 64;
  const int qw   = q0 + w * 16;

  const __bf16* qbase = Q  + (size_t)bh * 2048 * 128;
  const __bf16* kbase = Kb + (size_t)bh * 2048 * 128;
  const __bf16* vbase = Vb + (size_t)bh * 2048 * 128;

  // Q fragments live in registers for the whole kv loop (4 d-chunks of 32)
  v16bf qf[4];
  {
    const __bf16* qrow = qbase + (size_t)(qw + ln) * 128;
    for (int dc = 0; dc < 4; ++dc)
      qf[dc] = load_frag2(qrow + dc * 32 + 8 * g, qrow + dc * 32 + 16 + 8 * g);
  }

  v8f o[8];
  float mrow[8], lrow[8];
  for (int d = 0; d < 8; ++d) o[d] = vzero8();
  for (int i = 0; i < 8; ++i) { mrow[i] = -3e30f; lrow[i] = 0.0f; }

  const float scale = 0.08838834764831845f;  // 1/sqrt(128)

  for (int t = 0; t <= (int)blockIdx.y; ++t) {
    const int kv0 = t * 64;
    // ---- stage K tile (64x128) row-major and V tile transposed (d-major) ----
    for (int j = 0; j < 8; ++j) {
      const int idx  = j * 128 + tid;    // 0..1023 chunks of 8 bf16
      const int row  = idx >> 4;         // 0..63
      const int coff = (idx & 15) << 3;  // 0..120
      *(bf16x8*)&Ks[row][coff] =
          *(const bf16x8*)(kbase + (size_t)(kv0 + row) * 128 + coff);
      const bf16x8 vv =
          *(const bf16x8*)(vbase + (size_t)(kv0 + row) * 128 + coff);
      for (int e = 0; e < 8; ++e) Vts[coff + e][row] = vv[e];
    }
    __syncthreads();

    // ---- S = Q K^T  (16 x 64), 4 key-frags x 4 d-chunks ----
    v8f s[4];
    for (int nf = 0; nf < 4; ++nf) {
      v8f a = vzero8();
      const int kl = nf * 16 + ln;                 // B-frag: n = key = lane&15
      for (int dc = 0; dc < 4; ++dc) {
        const v16bf kf = load_frag2(&Ks[kl][dc * 32 + 8 * g],
                                    &Ks[kl][dc * 32 + 16 + 8 * g]);
        a = __builtin_amdgcn_wmma_f32_16x16x32_bf16(
                false, qf[dc], false, kf, (short)0, a, false, false);
      }
      s[nf] = a;
    }

    // ---- causal mask + online softmax (rows live on 16-lane half groups) ----
    float corr[8];
    for (int i = 0; i < 8; ++i) {
      const int qrow = qw + i + 8 * g;
      float rmax = -3e30f;
      for (int nf = 0; nf < 4; ++nf) {
        const int key = kv0 + nf * 16 + ln;
        float v = s[nf][i] * scale;
        v = (key <= qrow) ? v : -3e30f;
        s[nf][i] = v;
        rmax = fmaxf(rmax, v);
      }
      for (int msk = 8; msk >= 1; msk >>= 1)
        rmax = fmaxf(rmax, __shfl_xor(rmax, msk, 32));
      const float mnew = fmaxf(mrow[i], rmax);
      const float c = __expf(mrow[i] - mnew);
      float rsum = 0.0f;
      for (int nf = 0; nf < 4; ++nf) {
        const float p = __expf(s[nf][i] - mnew);
        s[nf][i] = p;
        rsum += p;
      }
      for (int msk = 8; msk >= 1; msk >>= 1)
        rsum += __shfl_xor(rsum, msk, 32);
      lrow[i] = lrow[i] * c + rsum;
      mrow[i] = mnew;
      corr[i] = c;
    }
    for (int df = 0; df < 8; ++df)
      for (int i = 0; i < 8; ++i) o[df][i] *= corr[i];

    // ---- P -> LDS (re-layout C-frag -> A-frag), then O += P V ----
    for (int nf = 0; nf < 4; ++nf)
      for (int i = 0; i < 8; ++i)
        Ps[w][i + 8 * g][nf * 16 + ln] = f2bf(s[nf][i]);
    __syncthreads();

    for (int df = 0; df < 8; ++df) {
      for (int kc = 0; kc < 2; ++kc) {
        const v16bf pf = load_frag2(&Ps[w][ln][kc * 32 + 8 * g],
                                    &Ps[w][ln][kc * 32 + 16 + 8 * g]);
        const v16bf vf = load_frag2(&Vts[df * 16 + ln][kc * 32 + 8 * g],
                                    &Vts[df * 16 + ln][kc * 32 + 16 + 8 * g]);
        o[df] = __builtin_amdgcn_wmma_f32_16x16x32_bf16(
                    false, pf, false, vf, (short)0, o[df], false, false);
      }
    }
    __syncthreads();
  }

  // ---- normalize and scatter to attn_out bf16 [B*T][2048] ----
  const int b = bh >> 4, h = bh & 15;
  for (int df = 0; df < 8; ++df) {
    for (int i = 0; i < 8; ++i) {
      const int qrow = qw + i + 8 * g;
      const int d = df * 16 + ln;
      const float val = o[df][i] / lrow[i];
      O[(size_t)(b * 2048 + qrow) * 2048 + h * 128 + d] = f2bf(val);
    }
  }
}

// ---------------------------------------------------------------------------
extern "C" void kernel_launch(void* const* d_in, const int* in_sizes, int n_in,
                              void* d_out, int out_size, void* d_ws, size_t ws_size,
                              hipStream_t stream) {
  (void)in_sizes; (void)n_in; (void)out_size; (void)ws_size;
  const float* x      = (const float*)d_in[0];
  const float* w_qkv  = (const float*)d_in[1];
  const float* w_proj = (const float*)d_in[2];
  const float* w_ff1  = (const float*)d_in[3];
  const float* b_ff1  = (const float*)d_in[4];
  const float* w_ff2  = (const float*)d_in[5];
  const float* b_ff2  = (const float*)d_in[6];
  float* out = (float*)d_out;

  char* ws = (char*)d_ws;
  const size_t MB = 1024ull * 1024ull;
  __bf16* qb   = (__bf16*)(ws);              // 16 MB  [B][H][T][128]
  __bf16* kb   = (__bf16*)(ws + 16 * MB);    // 16 MB
  __bf16* vb   = (__bf16*)(ws + 32 * MB);    // 16 MB
  __bf16* attn = (__bf16*)(ws + 48 * MB);    // 16 MB  [4096][2048]
  float*  proj = (float*) (ws + 64 * MB);    // 32 MB  [4096][2048]
  __bf16* hbuf = (__bf16*)(ws + 96 * MB);    // 64 MB  [4096][8192]

  dim3 blk(256);
  // 1) QKV projection: x[4096x2048] * w_qkv[2048x6144] -> q,k,v (bf16, per-head)
  gemm_wmma<0, float><<<dim3(48, 32), blk, 0, stream>>>(
      x, w_qkv, 6144, 2048, qb, kb, vb,
      nullptr, nullptr, nullptr, nullptr, nullptr, nullptr);
  // 2) causal flash attention -> attn[4096x2048] bf16
  attn_wmma<<<dim3(32, 32), dim3(128), 0, stream>>>(qb, kb, vb, attn);
  // 3) output projection: attn * w_proj -> proj (f32)
  gemm_wmma<1, __bf16><<<dim3(16, 32), blk, 0, stream>>>(
      attn, w_proj, 2048, 2048, nullptr, nullptr, nullptr,
      proj, nullptr, nullptr, nullptr, nullptr, nullptr);
  // 4) FF1 + exact GELU: x * w_ff1 + b_ff1 -> h (bf16)
  gemm_wmma<2, float><<<dim3(64, 32), blk, 0, stream>>>(
      x, w_ff1, 8192, 2048, nullptr, nullptr, nullptr,
      nullptr, hbuf, b_ff1, nullptr, nullptr, nullptr);
  // 5) FF2 + residuals: out = x + proj + h*w_ff2 + b_ff2
  gemm_wmma<3, __bf16><<<dim3(16, 32), blk, 0, stream>>>(
      hbuf, w_ff2, 2048, 8192, nullptr, nullptr, nullptr,
      nullptr, nullptr, b_ff2, x, proj, out);
}
